// CrossScanAttention_86457691668846
// MI455X (gfx1250) — matrix operand order repaired and checked by reading, not compiled, not run on hardware
//
#include <hip/hip_runtime.h>
#include <hip/hip_bf16.h>

// ---------------------------------------------------------------------------
// CrossScanAttention (bidirectional Mamba over pooled image channels), MI455X.
//
// Roofline: img1+img2 = 128 MB read dominates (≈5.5us @ 23.3 TB/s). All other
// stages touch ~50 MB of scratch and ~1 GFLOP. Strategy:
//   K1 pool     : bandwidth-bound max over H*W=1024 per (img,b,c)
//   K2 prep     : affine + layernorm -> f16 tokens (backward branch pre-reversed)
//   K3 in-GEMM  : xz = u @ in_w^T (M=16K, N=128, K=64)  -> v_wmma f32_16x16x32_f16
//   K4 conv     : causal depthwise conv K=4 + SiLU
//   K5 projGEMM : xdb = xc @ xproj^T (N=20 pad 32) and dt GEMM (K=4 pad 32) -> WMMA
//   K6 scan     : selective scan, parallel over (branch,b)x(d,n)=512 thr/block,
//                 chunked LDS staging + global_prefetch
//   K7 out      : rmsnorm + collapsed projection (out_w,post_w fold to one vec)
// ---------------------------------------------------------------------------

typedef __attribute__((ext_vector_type(16))) _Float16 v16h;
typedef __attribute__((ext_vector_type(8)))  float    v8f;

#define NB     16      // batch
#define LSEQ   1024    // sequence length (= C)
#define DM     64      // d_model / d_inner
#define EXZ    128     // 2*d_inner
#define NST    8       // d_state
#define NRANK  4       // dt_rank
#define NK20   20      // dt_rank + 2*d_state

// ---- WMMA fragment loaders (layouts per CDNA5 ISA 7.12.2) -------------------

// A matrix 16x32 f16 (MxK): lane L -> M=L%16, g=L/16; elem j -> K=(j/8)*16+g*8+(j%8)
__device__ __forceinline__ v16h load_a_frag(const _Float16* S, int row, int ld,
                                            int kBase, int lane) {
  const int g = (lane >> 4) & 1;
  v16h a;
#pragma unroll
  for (int j = 0; j < 16; ++j) {
    const int k = kBase + ((j >> 3) << 4) + (g << 3) + (j & 7);
    a[j] = S[row * ld + k];
  }
  return a;
}

// B matrix 32x16 f16 (KxN), stored as W[n][k]: lane -> n=lane%16, g=lane/16;
// elem j -> K = (j/8)*16 + g*8 + (j%8)
__device__ __forceinline__ v16h load_b_frag(const _Float16* W, int nBase, int ld,
                                            int kBase, int lane) {
  const int g = (lane >> 4) & 1;
  const int n = lane & 15;
  v16h b;
#pragma unroll
  for (int j = 0; j < 16; ++j) {
    b[j] = W[(nBase + n) * ld + kBase + ((j >> 3) << 4) + (g << 3) + (j & 7)];
  }
  return b;
}

__device__ __forceinline__ float siluf(float x) {
  return x / (1.f + __expf(-x));
}
__device__ __forceinline__ float softplusf(float x) {
  return (x > 20.f) ? x : log1pf(__expf(x));
}

// ---- K1: max pool over H*W=1024, one block per (img,b,c) -------------------
__global__ __launch_bounds__(256) void k1_pool(const float* __restrict__ img1,
                                               const float* __restrict__ img2,
                                               float* __restrict__ pbuf) {
  __shared__ float red[8];
  const int bid = blockIdx.x;            // 0..32767
  const int im  = bid >> 14;
  const int bc  = bid & 16383;
  const float* src = (im ? img2 : img1) + (size_t)bc * 1024;
  const int tid = threadIdx.x;
  const float4 v = reinterpret_cast<const float4*>(src)[tid];
  float m = fmaxf(fmaxf(v.x, v.y), fmaxf(v.z, v.w));
#pragma unroll
  for (int off = 16; off; off >>= 1) m = fmaxf(m, __shfl_xor(m, off, 32));
  if ((tid & 31) == 0) red[tid >> 5] = m;
  __syncthreads();
  if (tid == 0) {
    float r = red[0];
#pragma unroll
    for (int i = 1; i < 8; ++i) r = fmaxf(r, red[i]);
    pbuf[bid] = r;
  }
}

// ---- K2: affine + layernorm -> f16 token features ---------------------------
// One wave32 per token; lane covers d and d+32. Backward branch pre-reversed.
__global__ __launch_bounds__(256) void k2_prep(const float* __restrict__ pbuf,
                                               const float* __restrict__ pre_w,
                                               const float* __restrict__ pre_b,
                                               const float* __restrict__ ln_g,
                                               const float* __restrict__ ln_b,
                                               _Float16* __restrict__ u) {
  const int tid  = threadIdx.x;
  const int wave = tid >> 5, lane = tid & 31;
  const int token = blockIdx.x * 8 + wave;       // 0..32767 (branch,b,l)
  const int branch = token >> 14;
  const int t = token & 16383;
  const int b = t >> 10, l = t & 1023;
  const int src_l = branch ? (1023 - l) : l;     // backward branch reversed
  const float p = pbuf[branch * 16384 + b * 1024 + src_l];
  const int d0 = lane, d1 = lane + 32;
  const float x0 = p * pre_w[d0] + pre_b[d0];
  const float x1 = p * pre_w[d1] + pre_b[d1];
  float s = x0 + x1;
#pragma unroll
  for (int off = 16; off; off >>= 1) s += __shfl_xor(s, off, 32);
  const float mean = s * (1.f / 64.f);
  float vv = (x0 - mean) * (x0 - mean) + (x1 - mean) * (x1 - mean);
#pragma unroll
  for (int off = 16; off; off >>= 1) vv += __shfl_xor(vv, off, 32);
  const float inv = rsqrtf(vv * (1.f / 64.f) + 1e-5f);
  u[(size_t)token * 64 + d0] = (_Float16)((x0 - mean) * inv * ln_g[d0] + ln_b[d0]);
  u[(size_t)token * 64 + d1] = (_Float16)((x1 - mean) * inv * ln_g[d1] + ln_b[d1]);
}

// ---- K3: xz = u @ in_w^T  via WMMA. Per block: 128 tokens x 128 e ----------
__global__ __launch_bounds__(256) void k3_in_gemm(const _Float16* __restrict__ u,
                                                  const float* __restrict__ in_w_f,
                                                  const float* __restrict__ in_b_w_f,
                                                  float* __restrict__ xbuf,
                                                  float* __restrict__ zsbuf) {
  __shared__ _Float16 Us[128 * 64];   // tokens x d
  __shared__ _Float16 Wt[128 * 64];   // e x d (f16)
  const int bid = blockIdx.x;          // 2*16*8 = 256
  const int branch = bid >> 7;
  const int r = bid & 127;
  const int b = r >> 3;
  const int lBase = (r & 7) * 128;
  const int bb = branch * 16 + b;
  const float* wsrc = branch ? in_b_w_f : in_w_f;
  const _Float16* usrc = u + ((size_t)bb * LSEQ + lBase) * 64;
  const int tid = threadIdx.x;
  for (int idx = tid; idx < 128 * 64; idx += 256) {
    Us[idx] = usrc[idx];
    Wt[idx] = (_Float16)wsrc[idx];
  }
  __syncthreads();
  const int wave = tid >> 5, lane = tid & 31;
  const int mrow = wave * 16 + (lane & 15);
  const int g = (lane >> 4) & 1;
  const v16h a0 = load_a_frag(Us, mrow, 64, 0, lane);
  const v16h a1 = load_a_frag(Us, mrow, 64, 32, lane);
  for (int nt = 0; nt < 8; ++nt) {
    const int nBase = nt * 16;
    const v16h b0 = load_b_frag(Wt, nBase, 64, 0, lane);
    const v16h b1 = load_b_frag(Wt, nBase, 64, 32, lane);
    v8f acc = {};
    acc = __builtin_amdgcn_wmma_f32_16x16x32_f16(false, a0, false, b0,
                                                 (short)0, acc, false, false);
    acc = __builtin_amdgcn_wmma_f32_16x16x32_f16(false, a1, false, b1,
                                                 (short)0, acc, false, false);
    const int e = nBase + (lane & 15);
#pragma unroll
    for (int v = 0; v < 8; ++v) {
      const int m = v + (g << 3);
      const int l = lBase + wave * 16 + m;
      const float val = acc[v];
      if (e < 64) {
        xbuf[((size_t)bb * 64 + e) * LSEQ + l] = val;            // raw x -> conv
      } else {
        zsbuf[((size_t)bb * 64 + (e - 64)) * LSEQ + l] = siluf(val); // silu(z)
      }
    }
  }
}

// ---- K4: causal depthwise conv (K=4) + SiLU --------------------------------
__global__ __launch_bounds__(256) void k4_conv(const float* __restrict__ xbuf,
                                               const float* __restrict__ cw,
                                               const float* __restrict__ cb,
                                               const float* __restrict__ cwb,
                                               const float* __restrict__ cbb,
                                               float* __restrict__ xc) {
  const int idx = blockIdx.x * 256 + threadIdx.x;   // 2*16*64*1024 = 2M
  const int l = idx & 1023;
  const int d = (idx >> 10) & 63;
  const int bb = idx >> 16;                          // branch*16+b
  const int branch = bb >> 4;
  const float* w = (branch ? cwb : cw) + d * 4;
  float acc = (branch ? cbb : cb)[d];
  const float* xrow = xbuf + ((size_t)(idx >> 10) << 10);
#pragma unroll
  for (int k = 0; k < 4; ++k) {
    const int li = l - 3 + k;
    if (li >= 0) acc += xrow[li] * w[k];
  }
  xc[idx] = siluf(acc);
}

// ---- K5: xdb = xc @ xproj^T (N=20 pad32) and dt = softplus(dtlr@dt_w^T + b),
//          both via WMMA (K=4 GEMM zero-padded to K=32) -----------------------
__global__ __launch_bounds__(256) void k5_proj(const float* __restrict__ xc,
                                               const float* __restrict__ xp_f,
                                               const float* __restrict__ xpb_f,
                                               const float* __restrict__ dtw_f,
                                               const float* __restrict__ dtwb_f,
                                               const float* __restrict__ dtb_f,
                                               const float* __restrict__ dtbb_f,
                                               float* __restrict__ xdb,
                                               float* __restrict__ dtbuf) {
  __shared__ _Float16 Xs[128 * 64];   // tokens x d (f16)
  __shared__ _Float16 Wp[32 * 64];    // k20 (padded to 32) x d
  __shared__ float    Dws[64 * 4];    // dt_w [d][r]
  __shared__ float    dtlr[128 * 4];  // per-token dt_lr
  const int bid = blockIdx.x;          // 256
  const int branch = bid >> 7;
  const int r = bid & 127;
  const int b = r >> 3;
  const int lBase = (r & 7) * 128;
  const int bb = branch * 16 + b;
  const float* xp  = branch ? xpb_f  : xp_f;
  const float* dtw = branch ? dtwb_f : dtw_f;
  const float* dtb = branch ? dtbb_f : dtb_f;
  const int tid = threadIdx.x;
  const float* xcg = xc + (size_t)bb * 64 * LSEQ;
  for (int idx = tid; idx < 128 * 64; idx += 256) {
    const int d = idx >> 7, l = idx & 127;           // coalesced over l
    Xs[l * 64 + d] = (_Float16)xcg[(size_t)d * LSEQ + lBase + l];
  }
  for (int idx = tid; idx < 32 * 64; idx += 256) {
    const int kk = idx >> 6, d = idx & 63;
    Wp[idx] = (kk < NK20) ? (_Float16)xp[kk * 64 + d] : (_Float16)0.f;
  }
  if (tid < 256) {
    for (int idx = tid; idx < 256; idx += 256) Dws[idx] = dtw[idx];
  }
  __syncthreads();
  const int wave = tid >> 5, lane = tid & 31;
  const int g = (lane >> 4) & 1;
  const int n = lane & 15;
  const int mrow = wave * 16 + n;
  const v16h a0 = load_a_frag(Xs, mrow, 64, 0, lane);
  const v16h a1 = load_a_frag(Xs, mrow, 64, 32, lane);
  // --- xdb GEMM: N = 32 (20 valid) ---
  for (int nt = 0; nt < 2; ++nt) {
    const int nBase = nt * 16;
    const v16h b0 = load_b_frag(Wp, nBase, 64, 0, lane);
    const v16h b1 = load_b_frag(Wp, nBase, 64, 32, lane);
    v8f acc = {};
    acc = __builtin_amdgcn_wmma_f32_16x16x32_f16(false, a0, false, b0,
                                                 (short)0, acc, false, false);
    acc = __builtin_amdgcn_wmma_f32_16x16x32_f16(false, a1, false, b1,
                                                 (short)0, acc, false, false);
    const int kk = nBase + n;
#pragma unroll
    for (int v = 0; v < 8; ++v) {
      const int m = v + (g << 3);
      const int row = wave * 16 + m;
      if (kk < NK20)
        xdb[((size_t)bb * LSEQ + lBase + row) * NK20 + kk] = acc[v];
      if (kk < NRANK)
        dtlr[row * 4 + kk] = acc[v];
    }
  }
  __syncthreads();
  // --- dt GEMM: A = dtlr (K=4 zero-padded to 32), B = dt_w^T, N = 64 ---
  v16h ad;
#pragma unroll
  for (int j = 0; j < 16; ++j) ad[j] = (_Float16)0.f;
  if (g == 0) {
#pragma unroll
    for (int j = 0; j < 4; ++j) ad[j] = (_Float16)dtlr[mrow * 4 + j];
  }
  for (int nt = 0; nt < 4; ++nt) {
    const int nBase = nt * 16;
    v16h bd;
#pragma unroll
    for (int j = 0; j < 16; ++j) bd[j] = (_Float16)0.f;
    if (g == 0) {
#pragma unroll
      for (int j = 0; j < 4; ++j) bd[j] = (_Float16)Dws[(nBase + n) * 4 + j];
    }
    v8f acc = {};
    acc = __builtin_amdgcn_wmma_f32_16x16x32_f16(false, ad, false, bd,
                                                 (short)0, acc, false, false);
    const int d = nBase + n;
    const float bias = dtb[d];
#pragma unroll
    for (int v = 0; v < 8; ++v) {
      const int m = v + (g << 3);
      const int token = lBase + wave * 16 + m;
      dtbuf[((size_t)bb * LSEQ + token) * 64 + d] = softplusf(acc[v] + bias);
    }
  }
}

// ---- K6: selective scan. Block per (branch,b); thread = (d,n) pair ----------
__global__ __launch_bounds__(512) void k6_scan(const float* __restrict__ xc,
                                               const float* __restrict__ zs,
                                               const float* __restrict__ xdb,
                                               const float* __restrict__ dtbuf,
                                               const float* __restrict__ Al,
                                               const float* __restrict__ Alb,
                                               const float* __restrict__ Dp,
                                               const float* __restrict__ Dpb,
                                               float* __restrict__ ybuf) {
  __shared__ float dts[64][64];
  __shared__ float xs[64][65];
  __shared__ float zss[64][65];
  __shared__ float Bs[64][8];
  __shared__ float Cs[64][8];
  const int bb = blockIdx.x;                 // 0..31
  const int branch = bb >> 4;
  const int tid = threadIdx.x;
  const int d = tid >> 3, n = tid & 7;
  const float A_dn = -__expf((branch ? Alb : Al)[d * 8 + n]);
  const float Dd = (branch ? Dpb : Dp)[d];
  const float* dtg  = dtbuf + (size_t)bb * LSEQ * 64;
  const float* xdbg = xdb   + (size_t)bb * LSEQ * NK20;
  const float* xcg  = xc    + (size_t)bb * 64 * LSEQ;
  const float* zsg  = zs    + (size_t)bb * 64 * LSEQ;
  float h = 0.f;
  for (int c0 = 0; c0 < LSEQ; c0 += 64) {
    __syncthreads();
    if (c0 + 64 < LSEQ) {   // gfx1250 global_prefetch of next chunk
      __builtin_prefetch(&dtg[(size_t)(c0 + 64) * 64], 0, 1);
      __builtin_prefetch(&xdbg[(size_t)(c0 + 64) * NK20], 0, 1);
    }
    for (int idx = tid; idx < 4096; idx += 512)
      dts[idx >> 6][idx & 63] = dtg[(size_t)(c0 + (idx >> 6)) * 64 + (idx & 63)];
    for (int idx = tid; idx < 4096; idx += 512) {
      const int dd = idx >> 6, li = idx & 63;
      xs[dd][li]  = xcg[(size_t)dd * LSEQ + c0 + li];
      zss[dd][li] = zsg[(size_t)dd * LSEQ + c0 + li];
    }
    for (int idx = tid; idx < 1024; idx += 512) {
      const int li = idx >> 4, kk = idx & 15;
      const float v = xdbg[(size_t)(c0 + li) * NK20 + NRANK + kk];
      if (kk < 8) Bs[li][kk] = v; else Cs[li][kk - 8] = v;
    }
    __syncthreads();
    for (int i = 0; i < 64; ++i) {
      const float dtv = dts[i][d];
      const float xv  = xs[d][i];
      h = __expf(dtv * A_dn) * h + dtv * Bs[i][n] * xv;
      float cc = h * Cs[i][n];
      cc += __shfl_xor(cc, 1, 8);
      cc += __shfl_xor(cc, 2, 8);
      cc += __shfl_xor(cc, 4, 8);
      if (n == 0) {
        float y = (cc + Dd * xv) * zss[d][i];
        const int l = c0 + i;
        const int lo = branch ? (1023 - l) : l;   // un-reverse backward branch
        ybuf[((size_t)bb * LSEQ + lo) * 64 + d] = y;
      }
    }
  }
}

// ---- K7: y = yf+yb; rmsnorm; out_w/post_w fold into one 64-vector ----------
__global__ __launch_bounds__(256) void k7_out(const float* __restrict__ ybuf,
                                              const float* __restrict__ norm_w,
                                              const float* __restrict__ out_w,
                                              const float* __restrict__ post_w,
                                              const float* __restrict__ post_b,
                                              float* __restrict__ out) {
  __shared__ float weff[64];
  const int tid = threadIdx.x;
  if (tid < 64) {
    float s = 0.f;
    for (int e = 0; e < 64; ++e) s += post_w[e] * out_w[e * 64 + tid];
    weff[tid] = s * norm_w[tid];
  }
  __syncthreads();
  const int wave = tid >> 5, lane = tid & 31;
  const int token = blockIdx.x * 8 + wave;       // b*1024 + l, 0..16383
  const float* yf = ybuf + (size_t)token * 64;
  const float* yb = ybuf + (size_t)(16384 + token) * 64;
  const float v0 = yf[lane] + yb[lane];
  const float v1 = yf[lane + 32] + yb[lane + 32];
  float ss = v0 * v0 + v1 * v1;
#pragma unroll
  for (int off = 16; off; off >>= 1) ss += __shfl_xor(ss, off, 32);
  const float inv = rsqrtf(ss * (1.f / 64.f) + 1e-5f);
  float dot = v0 * inv * weff[lane] + v1 * inv * weff[lane + 32];
#pragma unroll
  for (int off = 16; off; off >>= 1) dot += __shfl_xor(dot, off, 32);
  if (lane == 0) out[token] = 1.f / (1.f + __expf(-(dot + post_b[0])));
}

// ---------------------------------------------------------------------------
extern "C" void kernel_launch(void* const* d_in, const int* in_sizes, int n_in,
                              void* d_out, int out_size, void* d_ws, size_t ws_size,
                              hipStream_t stream) {
  const float* img1   = (const float*)d_in[0];
  const float* img2   = (const float*)d_in[1];
  const float* pre_w  = (const float*)d_in[2];
  const float* pre_b  = (const float*)d_in[3];
  const float* ln_g   = (const float*)d_in[4];
  const float* ln_b   = (const float*)d_in[5];
  const float* in_w   = (const float*)d_in[6];
  const float* in_b_w = (const float*)d_in[7];
  const float* conv_w = (const float*)d_in[8];
  const float* conv_b = (const float*)d_in[9];
  const float* convb_w= (const float*)d_in[10];
  const float* convb_b= (const float*)d_in[11];
  const float* xproj  = (const float*)d_in[12];
  const float* xprojb = (const float*)d_in[13];
  const float* dt_w   = (const float*)d_in[14];
  const float* dt_b   = (const float*)d_in[15];
  const float* dtb_w  = (const float*)d_in[16];
  const float* dtb_b  = (const float*)d_in[17];
  const float* A_log  = (const float*)d_in[18];
  const float* A_b_log= (const float*)d_in[19];
  const float* Dp     = (const float*)d_in[20];
  const float* Dp_b   = (const float*)d_in[21];
  const float* norm_w = (const float*)d_in[22];
  const float* out_w  = (const float*)d_in[23];
  const float* post_w = (const float*)d_in[24];
  const float* post_b = (const float*)d_in[25];

  float* ws = (float*)d_ws;
  float*     pbuf   = ws;                             // 2*16*1024        = 32768 f
  _Float16*  ubuf   = (_Float16*)(ws + 32768);        // 2*16*1024*64 h   = 1048576 f
  float*     xbuf   = ws + 32768 + 1048576;           // 2*16*64*1024     = 2097152 f
  float*     zsbuf  = xbuf  + 2097152;                // silu(z)
  float*     xcbuf  = zsbuf + 2097152;                // conv+silu output
  float*     xdbbuf = xcbuf + 2097152;                // 2*16*1024*20     = 655360 f
  float*     dtbuf  = xdbbuf + 655360;                // 2*16*1024*64
  float*     ybuf   = dtbuf + 2097152;                // branch outputs

  k1_pool<<<32768, 256, 0, stream>>>(img1, img2, pbuf);
  k2_prep<<<4096, 256, 0, stream>>>(pbuf, pre_w, pre_b, ln_g, ln_b, ubuf);
  k3_in_gemm<<<256, 256, 0, stream>>>(ubuf, in_w, in_b_w, xbuf, zsbuf);
  k4_conv<<<8192, 256, 0, stream>>>(xbuf, conv_w, conv_b, convb_w, convb_b, xcbuf);
  k5_proj<<<256, 256, 0, stream>>>(xcbuf, xproj, xprojb, dt_w, dtb_w, dt_b, dtb_b,
                                   xdbbuf, dtbuf);
  k6_scan<<<32, 512, 0, stream>>>(xcbuf, zsbuf, xdbbuf, dtbuf,
                                  A_log, A_b_log, Dp, Dp_b, ybuf);
  k7_out<<<2048, 256, 0, stream>>>(ybuf, norm_w, out_w, post_w, post_b,
                                   (float*)d_out);
}
